// Policy_49941879718368
// MI455X (gfx1250) — matrix operand order repaired and checked by reading, not compiled, not run on hardware
//
#include <hip/hip_runtime.h>
#include <hip/hip_bf16.h>

// ---------------- CDNA5 WMMA types ----------------
typedef __attribute__((ext_vector_type(16))) __bf16 v16bf;
typedef __attribute__((ext_vector_type(8)))  float  v8f;

#define Bq 128
#define Lq 512
#define Vq 4096
#define Eq 512
#define Hq 1024

__device__ __forceinline__ unsigned short f2bf(float f) {
  unsigned int u = __float_as_uint(f);
  u += 0x7FFFu + ((u >> 16) & 1u);   // round-to-nearest-even
  return (unsigned short)(u >> 16);
}

union ABFrag {
  v16bf v;
  uint4 q[2];
};

// B fragment (32x16 bf16) from pre-transposed weights BT[n][k], row stride ldb.
// Lane l: column n = l&15 ; K runs = k0 + 8*(l>>4) .. +7 and k0 + 16 + 8*(l>>4) .. +7.
__device__ __forceinline__ v16bf load_b_frag(const unsigned short* __restrict__ bt,
                                             int ldb, int n0, int k0, int lane) {
  const int n = lane & 15, hh = lane >> 4;
  const unsigned short* p = bt + (size_t)(n0 + n) * (size_t)ldb + k0 + 8 * hh;
  ABFrag f;
  f.q[0] = *(const uint4*)(p);
  f.q[1] = *(const uint4*)(p + 16);
  return f.v;
}

__device__ __forceinline__ v8f bf16_wmma(v16bf a, v16bf b, v8f c) {
  return __builtin_amdgcn_wmma_f32_16x16x32_bf16(false, a, false, b,
                                                 (short)0, c, false, false);
}

// ---------------- pack kernels ----------------
__global__ void pack_emb(const float* __restrict__ src,
                         unsigned short* __restrict__ dst, int n) {
  int i = blockIdx.x * blockDim.x + threadIdx.x;
  if (i < n) dst[i] = f2bf(src[i]);
}

// dst[n*K + k] = bf16(src[k*N + n])  (weights to [n][k] so B frags load contiguous K)
__global__ void pack_transpose(const float* __restrict__ src,
                               unsigned short* __restrict__ dst, int K, int N) {
  int idx = blockIdx.x * blockDim.x + threadIdx.x;
  if (idx < K * N) {
    int n = idx / K, k = idx - n * K;
    dst[idx] = f2bf(src[(size_t)k * N + n]);
  }
}

__global__ void zero_h(float* __restrict__ h, unsigned short* __restrict__ hB, int n) {
  int i = blockIdx.x * blockDim.x + threadIdx.x;
  if (i < n) { h[i] = 0.0f; hB[i] = 0; }
}

// ---------------- gates_i = gather(emb, shifted x) @ Wi + bi ----------------
// 4x2 register tiling: each wave owns a 64x32 output patch (8 WMMAs / k-step,
// 12 b128 loads / k-step -> ~2.7x the arithmetic intensity of 1-tile waves).
// grid (65536/64 = 1024, 3072/(8*32) = 12), block 256 (8 waves).
__global__ __launch_bounds__(256) void gemm_gates(
    const int* __restrict__ x, const unsigned short* __restrict__ embB,
    const unsigned short* __restrict__ WiT, const float* __restrict__ bi,
    float* __restrict__ gates) {
  const int lane = threadIdx.x & 31;
  const int wave = threadIdx.x >> 5;
  const int row0 = blockIdx.x * 64;
  const int col0 = (blockIdx.y * 8 + wave) * 32;
  const int nl = lane & 15, hh = lane >> 4;

  // fused shifted-token embedding gather: row r uses token x[r-1], start=V at t==0
  const unsigned short* arow[4];
#pragma unroll
  for (int mi = 0; mi < 4; ++mi) {
    const int r = row0 + 16 * mi + nl;
    const int t = r & (Lq - 1);
    const int tok = (t == 0) ? Vq : x[r - 1];
    arow[mi] = embB + (size_t)tok * Eq + 8 * hh;
  }

  const float bv0 = bi[col0 + nl];
  const float bv1 = bi[col0 + 16 + nl];
  v8f c[4][2];
#pragma unroll
  for (int mi = 0; mi < 4; ++mi)
#pragma unroll
    for (int i = 0; i < 8; ++i) { c[mi][0][i] = bv0; c[mi][1][i] = bv1; }

  for (int k = 0; k < Eq; k += 32) {
    ABFrag a[4];
#pragma unroll
    for (int mi = 0; mi < 4; ++mi) {
      a[mi].q[0] = *(const uint4*)(arow[mi] + k);
      a[mi].q[1] = *(const uint4*)(arow[mi] + k + 16);
    }
    v16bf b0 = load_b_frag(WiT, Eq, col0, k, lane);
    v16bf b1 = load_b_frag(WiT, Eq, col0 + 16, k, lane);
#pragma unroll
    for (int mi = 0; mi < 4; ++mi) {
      c[mi][0] = bf16_wmma(a[mi].v, b0, c[mi][0]);
      c[mi][1] = bf16_wmma(a[mi].v, b1, c[mi][1]);
    }
  }

#pragma unroll
  for (int mi = 0; mi < 4; ++mi) {
    float* out = gates + (size_t)(row0 + 16 * mi + 8 * hh) * (3 * Hq) + col0 + nl;
#pragma unroll
    for (int i = 0; i < 8; ++i) {
      out[(size_t)i * (3 * Hq)] = c[mi][0][i];
      out[(size_t)i * (3 * Hq) + 16] = c[mi][1][i];
    }
  }
}

// ---------------- one GRU step: gh = h@Wh ; fuse gate math ----------------
// grid (128/16 = 8, 1024/16 = 64), block 32 (1 wave). Each wave does the
// r/z/n tiles for one (batch16 x hidden16) patch so the update fuses locally.
// Latency-critical (512 serial steps): Wh bf16 (6.3 MB) lives in the 192 MB L2.
__global__ __launch_bounds__(32) void gru_step(
    const unsigned short* __restrict__ hBin, unsigned short* __restrict__ hBout,
    float* __restrict__ h, const unsigned short* __restrict__ WhT,
    const float* __restrict__ gates, const float* __restrict__ bhn,
    unsigned short* __restrict__ hsB, int t) {
  const int lane = threadIdx.x & 31;
  const int row0 = blockIdx.x * 16;
  const int j0 = blockIdx.y * 16;
  const int nl = lane & 15, hh = lane >> 4;

  v8f cr = {}, cz = {}, cn = {};
  const unsigned short* arow = hBin + (size_t)(row0 + nl) * Hq + 8 * hh;

  for (int k = 0; k < Hq; k += 32) {
    ABFrag a;
    a.q[0] = *(const uint4*)(arow + k);
    a.q[1] = *(const uint4*)(arow + k + 16);
    v16bf br = load_b_frag(WhT, Hq, j0, k, lane);
    v16bf bz = load_b_frag(WhT, Hq, Hq + j0, k, lane);
    v16bf bn = load_b_frag(WhT, Hq, 2 * Hq + j0, k, lane);
    cr = bf16_wmma(a.v, br, cr);
    cz = bf16_wmma(a.v, bz, cz);
    cn = bf16_wmma(a.v, bn, cn);
  }

  const int col = j0 + nl;
  const float bnv = bhn[col];
#pragma unroll
  for (int i = 0; i < 8; ++i) {
    const int rr = row0 + 8 * hh + i;                         // batch row 0..127
    const size_t girow = ((size_t)rr * Lq + t) * (size_t)(3 * Hq);
    const float ir  = gates[girow + col];
    const float iz  = gates[girow + Hq + col];
    const float in_ = gates[girow + 2 * Hq + col];
    const float hold = h[(size_t)rr * Hq + col];
    const float rg = 1.0f / (1.0f + __expf(-(ir + cr[i])));
    const float zg = 1.0f / (1.0f + __expf(-(iz + cz[i])));
    const float ng = tanhf(in_ + rg * (cn[i] + bnv));
    const float hnew = (1.0f - zg) * ng + zg * hold;
    h[(size_t)rr * Hq + col] = hnew;
    const unsigned short hb = f2bf(hnew);
    hBout[(size_t)rr * Hq + col] = hb;
    hsB[((size_t)rr * Lq + t) * Hq + col] = hb;
  }
}

// ---------------- logits = hs @ Wp + bp ----------------
// 4x2 register tiling, grid (65536/64 = 1024, 4096/(8*32) = 16), block 256.
__global__ __launch_bounds__(256) void gemm_proj(
    const unsigned short* __restrict__ hsB, const unsigned short* __restrict__ WpT,
    const float* __restrict__ bp, float* __restrict__ out) {
  const int lane = threadIdx.x & 31;
  const int wave = threadIdx.x >> 5;
  const int row0 = blockIdx.x * 64;
  const int col0 = (blockIdx.y * 8 + wave) * 32;
  const int nl = lane & 15, hh = lane >> 4;

  const unsigned short* arow = hsB + (size_t)(row0 + nl) * Hq + 8 * hh;
  const float bv0 = bp[col0 + nl];
  const float bv1 = bp[col0 + 16 + nl];
  v8f c[4][2];
#pragma unroll
  for (int mi = 0; mi < 4; ++mi)
#pragma unroll
    for (int i = 0; i < 8; ++i) { c[mi][0][i] = bv0; c[mi][1][i] = bv1; }

  for (int k = 0; k < Hq; k += 32) {
    __builtin_prefetch(arow + k + 64, 0, 1);
    ABFrag a[4];
#pragma unroll
    for (int mi = 0; mi < 4; ++mi) {
      const unsigned short* p = arow + (size_t)(16 * mi) * Hq + k;
      a[mi].q[0] = *(const uint4*)(p);
      a[mi].q[1] = *(const uint4*)(p + 16);
    }
    v16bf b0 = load_b_frag(WpT, Hq, col0, k, lane);
    v16bf b1 = load_b_frag(WpT, Hq, col0 + 16, k, lane);
#pragma unroll
    for (int mi = 0; mi < 4; ++mi) {
      c[mi][0] = bf16_wmma(a[mi].v, b0, c[mi][0]);
      c[mi][1] = bf16_wmma(a[mi].v, b1, c[mi][1]);
    }
  }

#pragma unroll
  for (int mi = 0; mi < 4; ++mi) {
    float* o = out + (size_t)(row0 + 16 * mi + 8 * hh) * Vq + col0 + nl;
#pragma unroll
    for (int i = 0; i < 8; ++i) {
      o[(size_t)i * Vq] = c[mi][0][i];
      o[(size_t)i * Vq + 16] = c[mi][1][i];
    }
  }
}

// ---------------- host ----------------
extern "C" void kernel_launch(void* const* d_in, const int* in_sizes, int n_in,
                              void* d_out, int out_size, void* d_ws, size_t ws_size,
                              hipStream_t stream) {
  const int*   x   = (const int*)d_in[0];
  const float* emb = (const float*)d_in[1];
  const float* Wi  = (const float*)d_in[2];
  const float* bi  = (const float*)d_in[3];
  const float* Wh  = (const float*)d_in[4];
  const float* bhn = (const float*)d_in[5];
  const float* Wp  = (const float*)d_in[6];
  const float* bp  = (const float*)d_in[7];
  float* out = (float*)d_out;

  char* ws = (char*)d_ws;
  size_t off = 0;
  auto alloc = [&](size_t bytes) -> void* {
    void* p = ws + off;
    off += (bytes + 255) & ~(size_t)255;
    return p;
  };
  unsigned short* embB = (unsigned short*)alloc((size_t)(Vq + 1) * Eq * 2);
  unsigned short* WiT  = (unsigned short*)alloc((size_t)3 * Hq * Eq * 2);
  unsigned short* WhT  = (unsigned short*)alloc((size_t)3 * Hq * Hq * 2);
  unsigned short* WpT  = (unsigned short*)alloc((size_t)Vq * Hq * 2);
  float*          gates= (float*)alloc((size_t)Bq * Lq * 3 * Hq * 4);
  float*          h    = (float*)alloc((size_t)Bq * Hq * 4);
  unsigned short* hB0  = (unsigned short*)alloc((size_t)Bq * Hq * 2);
  unsigned short* hB1  = (unsigned short*)alloc((size_t)Bq * Hq * 2);
  unsigned short* hsB  = (unsigned short*)alloc((size_t)Bq * Lq * Hq * 2);

  { int n = (Vq + 1) * Eq;
    pack_emb<<<(n + 255) / 256, 256, 0, stream>>>(emb, embB, n); }
  { int n = Eq * 3 * Hq;
    pack_transpose<<<(n + 255) / 256, 256, 0, stream>>>(Wi, WiT, Eq, 3 * Hq); }
  { int n = Hq * 3 * Hq;
    pack_transpose<<<(n + 255) / 256, 256, 0, stream>>>(Wh, WhT, Hq, 3 * Hq); }
  { int n = Hq * Vq;
    pack_transpose<<<(n + 255) / 256, 256, 0, stream>>>(Wp, WpT, Hq, Vq); }
  { int n = Bq * Hq;
    zero_h<<<(n + 255) / 256, 256, 0, stream>>>(h, hB0, n); }

  // all input-to-hidden projections in one big WMMA GEMM (fused gather)
  gemm_gates<<<dim3(1024, 12), 256, 0, stream>>>(x, embB, WiT, bi, gates);

  // sequential scan: 512 dependent steps, double-buffered bf16 hidden state
  for (int t = 0; t < Lq; ++t) {
    const unsigned short* hin = (t & 1) ? hB1 : hB0;
    unsigned short*      hout = (t & 1) ? hB0 : hB1;
    gru_step<<<dim3(8, 64), 32, 0, stream>>>(hin, hout, h, WhT, gates, bhn, hsB, t);
  }

  // output projection straight into d_out
  gemm_proj<<<dim3(1024, 16), 256, 0, stream>>>(hsB, WpT, bp, out);
}
